// EncoderLayer_64424509440106
// MI455X (gfx1250) — matrix operand order repaired and checked by reading, not compiled, not run on hardware
//
#include <hip/hip_runtime.h>
#include <hip/hip_bf16.h>

typedef __attribute__((ext_vector_type(16))) _Float16 v16h;
typedef __attribute__((ext_vector_type(8)))  _Float16 h8;
typedef __attribute__((ext_vector_type(8)))  float    v8f;
typedef __attribute__((ext_vector_type(4)))  _Float16 h4;

// Exact parameter types for the async-LDS builtin (signature revealed by clang
// diagnostic: pointer to 16B int vector, global/LDS address spaces).
typedef int v4i_t __attribute__((vector_size(16)));
typedef __attribute__((address_space(1))) v4i_t gv4i_t;
typedef __attribute__((address_space(3))) v4i_t lv4i_t;

#define D_MODEL 768
#define N_HEAD  12
#define HEAD_DIM 64
#define D_FF    3072
#define SEQ     2048
#define BATCH   2
#define NTOK    (BATCH * SEQ)

// ---------------- CDNA5 async global->LDS copy (ASYNCcnt-tracked DMA path) ----------------
#if __has_builtin(__builtin_amdgcn_global_load_async_to_lds_b128)
#define HAVE_ASYNC_LDS 1
#else
#define HAVE_ASYNC_LDS 0
#endif

__device__ __forceinline__ void async_b128(const _Float16* g, _Float16* l) {
#if HAVE_ASYNC_LDS
    __builtin_amdgcn_global_load_async_to_lds_b128((gv4i_t*)g, (lv4i_t*)l, 0, 0);
#else
    *reinterpret_cast<h8*>(l) = *reinterpret_cast<const h8*>(g);
#endif
}

__device__ __forceinline__ void wait_async() {
#if HAVE_ASYNC_LDS
#if __has_builtin(__builtin_amdgcn_s_wait_asynccnt)
    __builtin_amdgcn_s_wait_asynccnt(0);
#else
    asm volatile("s_wait_asynccnt 0x0" ::: "memory");
#endif
#endif
}

// ---------------- WMMA fragment helpers (wave32, v_wmma_f32_16x16x32_f16) ----------------
// A 16x32 f16 (row-major LDS): lane 0-15 -> M=lane, halves 0..15=K0..15;
// lane 16-31 -> M=lane-16, halves 0..15=K16..31.
__device__ __forceinline__ v16h frag_a_from_lds(const _Float16* base, int stride, int lane) {
    const _Float16* p = base + (lane & 15) * stride + ((lane >> 4) << 4);
    h8 lo = *reinterpret_cast<const h8*>(p);        // ds_load_b128
    h8 hi = *reinterpret_cast<const h8*>(p + 8);    // ds_load_b128
    return __builtin_shufflevector(lo, hi, 0, 1, 2, 3, 4, 5, 6, 7,
                                           8, 9, 10, 11, 12, 13, 14, 15);
}

// B 32x16 f16 fragment-major: [lane 0..31][16 halves] contiguous (512 halves/frag).
// lane L holds: N = (L&15), K = (L>>4)*16 + i.
__device__ __forceinline__ v16h frag_b_fm(const _Float16* base, int lane) {
    const _Float16* p = base + lane * 16;
    h8 lo = *reinterpret_cast<const h8*>(p);
    h8 hi = *reinterpret_cast<const h8*>(p + 8);
    return __builtin_shufflevector(lo, hi, 0, 1, 2, 3, 4, 5, 6, 7,
                                           8, 9, 10, 11, 12, 13, 14, 15);
}

__device__ __forceinline__ v8f zero8() {
    v8f z;
#pragma unroll
    for (int i = 0; i < 8; ++i) z[i] = 0.0f;
    return z;
}

// ---------------- Weight pre-pack: W[K,N] f32 -> f16 fragment-major tiles ----------------
// Tile (kt, nb) = 8 fragments x 32 lanes x 16 halves = 4096 halves = 8KB, contiguous.
// Element (j, L, i) = W[kt*32 + (L>>4)*16 + i][nb*128 + j*16 + (L&15)].
__global__ __launch_bounds__(256)
void pack_w_kernel(const float* __restrict__ W, _Float16* __restrict__ PW, int N) {
    const int kt = blockIdx.y, nb = blockIdx.x;
    const int L = threadIdx.x & 31;
    const int col  = nb * 128 + ((threadIdx.x >> 5) << 4) + (L & 15);
    const int krow = kt * 32 + ((L >> 4) << 4);
    h8 lo, hi;
#pragma unroll
    for (int i = 0; i < 8; ++i) lo[i] = (_Float16)W[(size_t)(krow + i) * N + col];
#pragma unroll
    for (int i = 0; i < 8; ++i) hi[i] = (_Float16)W[(size_t)(krow + 8 + i) * N + col];
    _Float16* dst = PW + ((((size_t)kt * gridDim.x + nb) << 8) + threadIdx.x) * 16;
    *reinterpret_cast<h8*>(dst)     = lo;
    *reinterpret_cast<h8*>(dst + 8) = hi;
}

// ---------------- GEMM: C = act(A[M,K] @ B[K,N] + bias (+ residual)) ----------------
// A: f32 row-major (register-staged f32->f16). B: pre-packed f16 tiles, fetched with
// async global->LDS b128 copies. Double-buffered LDS, one barrier per k-step.
#define BM 128
#define BN 128
#define BK 32
#define LDSA_STRIDE (BK + 8)   // 40 halves/row (80B, 16B-aligned rows)

__global__ __launch_bounds__(256)
void gemm_kernel(const float* __restrict__ A, const _Float16* __restrict__ PB,
                 const float* __restrict__ bias, const float* __restrict__ residual,
                 float* __restrict__ C, int M, int N, int K, int gelu_act) {
    __shared__ __align__(16) _Float16 lA[2][BM * LDSA_STRIDE];
    __shared__ __align__(32) _Float16 lB[2][8 * 32 * 16];
    const int tid  = threadIdx.x;
    const int lane = tid & 31;
    const int wave = tid >> 5;
    const int bm = blockIdx.y * BM;
    const int nblk = gridDim.x;            // N / 128
    const int wm = (wave >> 1) * 32;
    const int wn = (wave & 1) * 64;

    v8f acc[2][4];
#pragma unroll
    for (int i = 0; i < 2; ++i)
#pragma unroll
        for (int j = 0; j < 4; ++j) acc[i][j] = zero8();

    float4 ra[4];
    auto g_loadA = [&](int k0) {
#pragma unroll
        for (int it = 0; it < 4; ++it) {
            int idx = tid + it * 256;
            ra[it] = *reinterpret_cast<const float4*>(
                A + (size_t)(bm + (idx >> 3)) * K + k0 + ((idx & 7) << 2));
        }
    };
    auto storeA = [&](int buf) {
#pragma unroll
        for (int it = 0; it < 4; ++it) {
            int idx = tid + it * 256;
            int r = idx >> 3, c4 = (idx & 7) << 2;
            h4 hh; hh[0] = (_Float16)ra[it].x; hh[1] = (_Float16)ra[it].y;
                   hh[2] = (_Float16)ra[it].z; hh[3] = (_Float16)ra[it].w;
            *reinterpret_cast<h4*>(&lA[buf][r * LDSA_STRIDE + c4]) = hh;
        }
    };
    auto asyncB = [&](int kt, int buf) {   // copy one 8KB packed tile (2 x b128 per lane)
        const _Float16* src = PB + (((size_t)kt * nblk + blockIdx.x) << 12) + (tid << 4);
        _Float16* dst = &lB[buf][tid << 4];
        async_b128(src, dst);
        async_b128(src + 8, dst + 8);
    };

    g_loadA(0);
    asyncB(0, 0);
    storeA(0);
    wait_async();
    __syncthreads();

    int buf = 0;
    for (int k0 = 0;;) {
        const bool more = (k0 + BK) < K;
        if (more) {
            g_loadA(k0 + BK);              // overlaps WMMA below
            asyncB((k0 + BK) >> 5, buf ^ 1);
        }

        v16h af[2], bf[4];
#pragma unroll
        for (int i = 0; i < 2; ++i)
            af[i] = frag_a_from_lds(&lA[buf][(wm + i * 16) * LDSA_STRIDE], LDSA_STRIDE, lane);
#pragma unroll
        for (int j = 0; j < 4; ++j)
            bf[j] = frag_b_fm(&lB[buf][((wn >> 4) + j) << 9], lane);
#pragma unroll
        for (int i = 0; i < 2; ++i)
#pragma unroll
            for (int j = 0; j < 4; ++j)
                acc[i][j] = __builtin_amdgcn_wmma_f32_16x16x32_f16(
                    false, af[i], false, bf[j], (short)0, acc[i][j], false, false);

        if (!more) break;
        storeA(buf ^ 1);                   // other buffer: no barrier needed before store
        wait_async();
        __syncthreads();                   // single barrier per k-step
        buf ^= 1;
        k0 += BK;
    }

    // Epilogue: C layout -> lane holds column (lane&15), vgpr r -> row r + (lane>>4)*8
    const int bn = blockIdx.x * BN;
    const int ml = (lane >> 4) * 8;
    const int nl = lane & 15;
#pragma unroll
    for (int i = 0; i < 2; ++i) {
#pragma unroll
        for (int j = 0; j < 4; ++j) {
#pragma unroll
            for (int r = 0; r < 8; ++r) {
                int row = bm + wm + i * 16 + ml + r;
                int col = bn + wn + j * 16 + nl;
                float v = acc[i][j][r] + bias[col];
                if (residual) v += residual[(size_t)row * N + col];
                if (gelu_act) v = 0.5f * v * (1.0f + erff(v * 0.70710678118654752f));
                C[(size_t)row * N + col] = v;
            }
        }
    }
}

// ---------------- Flash attention: per (batch, head, 128-query tile) ----------------
#define QT 128
#define KT 128
#define QSTR (HEAD_DIM + 8)   // 72 halves (144B rows, 16B-aligned)
#define PSTR (KT + 8)         // 136 halves (272B rows, 16B-aligned)

__global__ __launch_bounds__(256)
void attention_kernel(const float* __restrict__ Q, const float* __restrict__ Kb,
                      const float* __restrict__ Vb, float* __restrict__ O) {
    __shared__ __align__(16) _Float16 lQ[QT * QSTR];         // row-major A-side
    __shared__ __align__(32) _Float16 lKf[2 * 8 * 32 * 16];  // K^T frags: [kk2][j][lane][16]
    __shared__ __align__(32) _Float16 lVf[4 * 4 * 32 * 16];  // V frags:   [kk2][t][lane][16]
    __shared__ __align__(16) _Float16 lP[8 * 16 * PSTR];     // per-wave 16x128 P (A-side)
    const int tid  = threadIdx.x;
    const int lane = tid & 31;
    const int wave = tid >> 5;
    const int qbase = blockIdx.x * QT;
    const int h  = blockIdx.y;
    const int bb = blockIdx.z;
    const size_t tok0 = (size_t)bb * SEQ;

    // Q tile (pre-scaled by 1/sqrt(64)) : 128x64 f32 -> f16 row-major
#pragma unroll
    for (int it = 0; it < 8; ++it) {
        int idx = tid + it * 256;
        int r = idx >> 4, c4 = (idx & 15) << 2;
        float4 v = *reinterpret_cast<const float4*>(
            Q + (tok0 + qbase + r) * D_MODEL + h * HEAD_DIM + c4);
        h4 hh; hh[0] = (_Float16)(v.x * 0.125f); hh[1] = (_Float16)(v.y * 0.125f);
               hh[2] = (_Float16)(v.z * 0.125f); hh[3] = (_Float16)(v.w * 0.125f);
        *reinterpret_cast<h4*>(&lQ[r * QSTR + c4]) = hh;
    }

    float m_st[8], l_st[8];
#pragma unroll
    for (int r = 0; r < 8; ++r) { m_st[r] = -1e30f; l_st[r] = 0.0f; }
    v8f oacc[4];
#pragma unroll
    for (int t = 0; t < 4; ++t) oacc[t] = zero8();

    _Float16* myP = &lP[wave * 16 * PSTR];
    const int ml = (lane >> 4) * 8;
    const int nl = lane & 15;

    for (int kt = 0; kt < SEQ / KT; ++kt) {
        __syncthreads();   // protect lKf/lVf reuse; also orders initial lQ fill
#pragma unroll
        for (int it = 0; it < 8; ++it) {
            int idx = tid + it * 256;
            int r = idx >> 4, c4 = (idx & 15) << 2;   // r = key row, c4 = head-dim cols
            const size_t base = (tok0 + (size_t)kt * KT + r) * D_MODEL + h * HEAD_DIM + c4;
            float4 kv = *reinterpret_cast<const float4*>(Kb + base);
            float4 vv = *reinterpret_cast<const float4*>(Vb + base);
            {   // K^T fragment-major: fast global axis (kd) == fragment half index -> h4 store
                int kk2 = c4 >> 5, hb = (c4 >> 4) & 1;
                int L = (r & 15) + (hb << 4), j = r >> 4, i0 = c4 & 15;
                h4 hh; hh[0] = (_Float16)kv.x; hh[1] = (_Float16)kv.y;
                       hh[2] = (_Float16)kv.z; hh[3] = (_Float16)kv.w;
                *reinterpret_cast<h4*>(&lKf[(((((kk2 << 3) + j) << 5) + L) << 4) + i0]) = hh;
            }
            {   // V fragment-major: lane index varies across the 4 columns -> scalar scatter
                int kk2 = r >> 5, hb = (r >> 4) & 1;
                int t = c4 >> 4, Lb = (c4 & 15) + (hb << 4);
                _Float16* p = &lVf[(((kk2 << 2) + t) << 9) + (r & 15)];
                p[(Lb + 0) << 4] = (_Float16)vv.x;
                p[(Lb + 1) << 4] = (_Float16)vv.y;
                p[(Lb + 2) << 4] = (_Float16)vv.z;
                p[(Lb + 3) << 4] = (_Float16)vv.w;
            }
        }
        __syncthreads();

        // S = Q(16x64) @ K^T(64x128): 8 column tiles, 2 k-steps
        v8f sfr[8];
#pragma unroll
        for (int j = 0; j < 8; ++j) sfr[j] = zero8();
#pragma unroll
        for (int kk2 = 0; kk2 < 2; ++kk2) {
            v16h a = frag_a_from_lds(&lQ[(wave * 16) * QSTR + kk2 * 32], QSTR, lane);
#pragma unroll
            for (int j = 0; j < 8; ++j) {
                v16h b = frag_b_fm(&lKf[((kk2 << 3) + j) << 9], lane);
                sfr[j] = __builtin_amdgcn_wmma_f32_16x16x32_f16(
                    false, a, false, b, (short)0, sfr[j], false, false);
            }
        }

        // Online softmax; row = r + (lane>>4)*8, columns striped across 16-lane halves
#pragma unroll
        for (int r = 0; r < 8; ++r) {
            float tmax = -1e30f;
#pragma unroll
            for (int j = 0; j < 8; ++j) tmax = fmaxf(tmax, sfr[j][r]);
#pragma unroll
            for (int m = 1; m < 16; m <<= 1) tmax = fmaxf(tmax, __shfl_xor(tmax, m, 32));
            float mnew = fmaxf(m_st[r], tmax);
            float corr = __expf(m_st[r] - mnew);
            float rsum = 0.0f;
#pragma unroll
            for (int j = 0; j < 8; ++j) {
                float p = __expf(sfr[j][r] - mnew);
                sfr[j][r] = p;
                rsum += p;
            }
#pragma unroll
            for (int m = 1; m < 16; m <<= 1) rsum += __shfl_xor(rsum, m, 32);
            l_st[r] = l_st[r] * corr + rsum;
            m_st[r] = mnew;
#pragma unroll
            for (int t = 0; t < 4; ++t) oacc[t][r] *= corr;
        }

        // Stage P (f16) into wave-private LDS (A-side row-major), re-read as A fragments
#pragma unroll
        for (int j = 0; j < 8; ++j)
#pragma unroll
            for (int r = 0; r < 8; ++r)
                myP[(ml + r) * PSTR + j * 16 + nl] = (_Float16)sfr[j][r];

        // O += P(16x128) @ V(128x64)
#pragma unroll
        for (int kk2 = 0; kk2 < 4; ++kk2) {
            v16h a = frag_a_from_lds(&myP[kk2 * 32], PSTR, lane);
#pragma unroll
            for (int t = 0; t < 4; ++t) {
                v16h b = frag_b_fm(&lVf[((kk2 << 2) + t) << 9], lane);
                oacc[t] = __builtin_amdgcn_wmma_f32_16x16x32_f16(
                    false, a, false, b, (short)0, oacc[t], false, false);
            }
        }
    }

    // Normalize and write ctx in [tok, h*64+col] layout
#pragma unroll
    for (int t = 0; t < 4; ++t)
#pragma unroll
        for (int r = 0; r < 8; ++r) {
            size_t token = tok0 + qbase + wave * 16 + ml + r;
            O[token * D_MODEL + h * HEAD_DIM + t * 16 + nl] = oacc[t][r] / l_st[r];
        }
}

// ---------------- LayerNorm: one block per token row (768 = 3*256) ----------------
__global__ __launch_bounds__(256)
void layernorm_kernel(const float* __restrict__ in, const float* __restrict__ g,
                      const float* __restrict__ b, float* __restrict__ out) {
    __shared__ float redS[8], redQ[8];
    const int row = blockIdx.x;
    const int tid = threadIdx.x;
    const float* x = in + (size_t)row * D_MODEL;
    float v[3], s = 0.0f, ss = 0.0f;
#pragma unroll
    for (int i = 0; i < 3; ++i) {
        v[i] = x[tid + i * 256];
        s += v[i]; ss += v[i] * v[i];
    }
#pragma unroll
    for (int m = 1; m < 32; m <<= 1) { s += __shfl_xor(s, m, 32); ss += __shfl_xor(ss, m, 32); }
    const int wave = tid >> 5, lane = tid & 31;
    if (lane == 0) { redS[wave] = s; redQ[wave] = ss; }
    __syncthreads();
    if (tid == 0) {
        float a = 0.0f, q = 0.0f;
        for (int i = 0; i < 8; ++i) { a += redS[i]; q += redQ[i]; }
        redS[0] = a; redQ[0] = q;
    }
    __syncthreads();
    const float mu  = redS[0] * (1.0f / 768.0f);
    const float var = redQ[0] * (1.0f / 768.0f) - mu * mu;
    const float inv = rsqrtf(var + 1e-5f);
#pragma unroll
    for (int i = 0; i < 3; ++i) {
        int c = tid + i * 256;
        out[(size_t)row * D_MODEL + c] = (v[i] - mu) * inv * g[c] + b[c];
    }
}

// ---------------- Launcher ----------------
extern "C" void kernel_launch(void* const* d_in, const int* in_sizes, int n_in,
                              void* d_out, int out_size, void* d_ws, size_t ws_size,
                              hipStream_t stream) {
    const float* x   = (const float*)d_in[0];
    const float* Wq  = (const float*)d_in[1];
    const float* bq  = (const float*)d_in[2];
    const float* Wk  = (const float*)d_in[3];
    const float* bk  = (const float*)d_in[4];
    const float* Wv  = (const float*)d_in[5];
    const float* bv  = (const float*)d_in[6];
    const float* Wo  = (const float*)d_in[7];
    const float* bo  = (const float*)d_in[8];
    const float* W1  = (const float*)d_in[9];
    const float* b1  = (const float*)d_in[10];
    const float* W2  = (const float*)d_in[11];
    const float* b2  = (const float*)d_in[12];
    const float* g1  = (const float*)d_in[13];
    const float* be1 = (const float*)d_in[14];
    const float* g2  = (const float*)d_in[15];
    const float* be2 = (const float*)d_in[16];
    (void)in_sizes; (void)n_in; (void)out_size; (void)ws_size;

    float* ws  = (float*)d_ws;
    const size_t TOKD = (size_t)NTOK * D_MODEL;          // 3,145,728
    const size_t TOKF = (size_t)NTOK * D_FF;             // 12,582,912
    float* Qb  = ws;                // reused as y1
    float* Kbf = ws + TOKD;         // reused as h
    float* Vbf = ws + 2 * TOKD;     // reused as y2
    float* ctx = ws + 3 * TOKD;
    float* ff  = ws + 4 * TOKD;     // 4096 x 3072
    float* out = (float*)d_out;

    // Packed f16 weights (fragment-major tiles) after the f32 scratch region.
    const size_t PSQ = (size_t)(D_MODEL / 32) * (D_MODEL / 128) * 4096;   // 589,824 halves
    const size_t PS1 = (size_t)(D_MODEL / 32) * (D_FF / 128) * 4096;      // 2,359,296
    const size_t PS2 = (size_t)(D_FF / 32) * (D_MODEL / 128) * 4096;      // 2,359,296
    _Float16* pk  = (_Float16*)(ws + 4 * TOKD + TOKF);
    _Float16* PWq = pk;
    _Float16* PWk = pk + PSQ;
    _Float16* PWv = pk + 2 * PSQ;
    _Float16* PWo = pk + 3 * PSQ;
    _Float16* PW1 = pk + 4 * PSQ;
    _Float16* PW2 = PW1 + PS1;
    (void)PS2;

    dim3 blk(256);
    dim3 gpq(D_MODEL / 128, D_MODEL / 32);   // (6, 24)
    dim3 gp1(D_FF / 128,    D_MODEL / 32);   // (24, 24)
    dim3 gp2(D_MODEL / 128, D_FF / 32);      // (6, 96)
    pack_w_kernel<<<gpq, blk, 0, stream>>>(Wq, PWq, D_MODEL);
    pack_w_kernel<<<gpq, blk, 0, stream>>>(Wk, PWk, D_MODEL);
    pack_w_kernel<<<gpq, blk, 0, stream>>>(Wv, PWv, D_MODEL);
    pack_w_kernel<<<gpq, blk, 0, stream>>>(Wo, PWo, D_MODEL);
    pack_w_kernel<<<gp1, blk, 0, stream>>>(W1, PW1, D_FF);
    pack_w_kernel<<<gp2, blk, 0, stream>>>(W2, PW2, D_MODEL);

    dim3 g768(D_MODEL / BN, NTOK / BM);                  // (6, 32)
    gemm_kernel<<<g768, blk, 0, stream>>>(x, PWq, bq, nullptr, Qb,  NTOK, D_MODEL, D_MODEL, 0);
    gemm_kernel<<<g768, blk, 0, stream>>>(x, PWk, bk, nullptr, Kbf, NTOK, D_MODEL, D_MODEL, 0);
    gemm_kernel<<<g768, blk, 0, stream>>>(x, PWv, bv, nullptr, Vbf, NTOK, D_MODEL, D_MODEL, 0);

    dim3 ga(SEQ / QT, N_HEAD, BATCH);                    // (16, 12, 2)
    attention_kernel<<<ga, blk, 0, stream>>>(Qb, Kbf, Vbf, ctx);

    float* y1 = Qb;   // Q dead after attention
    gemm_kernel<<<g768, blk, 0, stream>>>(ctx, PWo, bo, x, y1, NTOK, D_MODEL, D_MODEL, 0);

    float* hbuf = Kbf; // K dead
    layernorm_kernel<<<dim3(NTOK), blk, 0, stream>>>(y1, g1, be1, hbuf);

    dim3 gff1(D_FF / BN, NTOK / BM);                     // (24, 32)
    gemm_kernel<<<gff1, blk, 0, stream>>>(hbuf, PW1, b1, nullptr, ff, NTOK, D_FF, D_MODEL, 1);

    float* y2 = Vbf;  // V dead
    gemm_kernel<<<g768, blk, 0, stream>>>(ff, PW2, b2, hbuf, y2, NTOK, D_MODEL, D_FF, 0);

    layernorm_kernel<<<dim3(NTOK), blk, 0, stream>>>(y2, g2, be2, out);
}
